// AFMLayer_87162066305261
// MI455X (gfx1250) — matrix operand order repaired and checked by reading, not compiled, not run on hardware
//
#include <hip/hip_runtime.h>
#include <stdint.h>

#define BATCH 4096
#define NF 50
#define ND 64
#define NA 64
#define NP 1225      // F*(F-1)/2
#define NPT 77       // ceil(NP/16)
#define NPPAD 1232   // NPT*16
#define STRIDE 68    // padded f32 row stride (68*4 B = 272 B, 16B-aligned, breaks 64-bank period)

typedef __attribute__((ext_vector_type(16))) __bf16 v16bf;
typedef __attribute__((ext_vector_type(8)))  float  v8f;

union BF16x16 { v16bf v; __bf16 h[16]; unsigned short u[16]; uint4 q[2]; };

__device__ __forceinline__ unsigned short f2bf(float f) {
  union { float f; unsigned int u; } x; x.f = f;
  unsigned int r = x.u + 0x7FFFu + ((x.u >> 16) & 1u);  // round-to-nearest-even
  return (unsigned short)(r >> 16);
}

__global__ __launch_bounds__(256)
void afm_kernel(const float* __restrict__ inp, const float* __restrict__ W1g,
                const float* __restrict__ b1g, const float* __restrict__ w2g,
                const float* __restrict__ pv,  float* __restrict__ out)
{
  __shared__ __align__(16) float          sInF[NF * STRIDE];   // f32 inputs[b], padded rows
  __shared__ __align__(16) unsigned short sW1T[NA * ND];       // bf16 W1 transposed: [n][k]
  __shared__ __align__(16) float          sLogit[NPPAD];
  __shared__ unsigned char                sI0[NPPAD], sI1[NPPAD];
  __shared__ __align__(16) float          sPool[16 * 64];      // per-group pooled partials
  __shared__ __align__(16) float          sRed[256];
  __shared__ float sM, sZ;

  const int b    = blockIdx.x;
  const int tid  = threadIdx.x;
  const int wave = tid >> 5;
  const int lane = tid & 31;

  // ---- Phase 0: stage inputs[b] (padded), pair tables, transposed bf16 W1 ----
  const float* gin = inp + (size_t)b * NF * ND;
  for (int idx = tid; idx < NF * ND; idx += 256) {
    int i = idx >> 6, k = idx & 63;
    sInF[i * STRIDE + k] = gin[idx];
  }

  for (int p = tid; p < NPPAD; p += 256) {
    int i = 0, j = 0;
    if (p < NP) {
      // p = i*(2F-i-1)/2 + (j-i-1), F=50 -> off(i) = i*(99-i)/2
      float disc = 9801.0f - 8.0f * (float)p;
      i = (int)floorf((99.0f - sqrtf(disc)) * 0.5f);
      if (i < 0) i = 0;
      while (i > 0 && i * (99 - i) / 2 > p) --i;
      while ((i + 1) * (99 - (i + 1)) / 2 <= p) ++i;
      j = p - i * (99 - i) / 2 + i + 1;
    }
    sI0[p] = (unsigned char)i; sI1[p] = (unsigned char)j;
  }

  for (int idx = tid; idx < NA * ND; idx += 256) {
    int n = idx >> 6, k = idx & 63;
    sW1T[n * ND + k] = f2bf(W1g[k * NA + n]);   // transpose: contiguous-K per column
  }
  __syncthreads();

  // ---- Phase 1: per-wave B fragments (all of W1) + per-lane bias/w2 ----
  const int nlo   = lane & 15;
  const int khalf = lane >> 4;          // 0: lanes 0-15, 1: lanes 16-31
  BF16x16 Bf[4][2];                     // [ntile][khalf-of-K64]
  #pragma unroll
  for (int nt = 0; nt < 4; ++nt)
    #pragma unroll
    for (int kk = 0; kk < 2; ++kk) {
      // B 32x16 layout: col n = lane%16; lanes 0-15 hold K 0-7,16-23, lanes 16-31 K 8-15,24-31
      const int base = (nt * 16 + nlo) * ND + kk * 32 + khalf * 8;
      Bf[nt][kk].q[0] = *reinterpret_cast<const uint4*>(&sW1T[base]);
      Bf[nt][kk].q[1] = *reinterpret_cast<const uint4*>(&sW1T[base + 16]);
    }
  float b1v[4], w2v[4];
  #pragma unroll
  for (int nt = 0; nt < 4; ++nt) { b1v[nt] = b1g[nt * 16 + nlo]; w2v[nt] = w2g[nt * 16 + nlo]; }

  const int kb = khalf * 8;             // A 16x32 layout K chunk base (0 or 8)

  // ---- Phase 2: prod (in registers) -> WMMA bf16 GEMM -> ReLU -> dot(w2) -> logits ----
  for (int t = wave; t < NPT; t += 8) {
    // Lane owns A row M = lane&15 -> pair p; build both K-half fragments directly.
    const int p  = t * 16 + nlo;                  // >= NP rows produce discarded garbage (finite)
    const int i0 = sI0[p], i1 = sI1[p];
    const float* r0 = &sInF[i0 * STRIDE];
    const float* r1 = &sInF[i1 * STRIDE];

    BF16x16 Af[2];
    #pragma unroll
    for (int kk = 0; kk < 2; ++kk) {
      #pragma unroll
      for (int c = 0; c < 2; ++c) {               // halves j=0..7 -> K kbase.., j=8..15 -> K kbase+16..
        const int kbase = kk * 32 + kb + c * 16;
        const float4 x0 = *reinterpret_cast<const float4*>(&r0[kbase]);
        const float4 x1 = *reinterpret_cast<const float4*>(&r0[kbase + 4]);
        const float4 y0 = *reinterpret_cast<const float4*>(&r1[kbase]);
        const float4 y1 = *reinterpret_cast<const float4*>(&r1[kbase + 4]);
        Af[kk].h[c * 8 + 0] = (__bf16)(x0.x * y0.x);
        Af[kk].h[c * 8 + 1] = (__bf16)(x0.y * y0.y);
        Af[kk].h[c * 8 + 2] = (__bf16)(x0.z * y0.z);
        Af[kk].h[c * 8 + 3] = (__bf16)(x0.w * y0.w);
        Af[kk].h[c * 8 + 4] = (__bf16)(x1.x * y1.x);
        Af[kk].h[c * 8 + 5] = (__bf16)(x1.y * y1.y);
        Af[kk].h[c * 8 + 6] = (__bf16)(x1.z * y1.z);
        Af[kk].h[c * 8 + 7] = (__bf16)(x1.w * y1.w);
      }
    }

    float lacc[8];
    #pragma unroll
    for (int r = 0; r < 8; ++r) lacc[r] = 0.0f;

    #pragma unroll
    for (int nt = 0; nt < 4; ++nt) {
      v8f c = {};
      c = __builtin_amdgcn_wmma_f32_16x16x32_bf16(false, Af[0].v, false, Bf[nt][0].v,
                                                  (short)0, c, false, false);
      c = __builtin_amdgcn_wmma_f32_16x16x32_bf16(false, Af[1].v, false, Bf[nt][1].v,
                                                  (short)0, c, false, false);
      // D layout: VGPR r = row (r + 8*khalf), lane%16 = col n = nt*16+nlo
      #pragma unroll
      for (int r = 0; r < 8; ++r) {
        float h = fmaxf(c[r] + b1v[nt], 0.0f);
        lacc[r] += h * w2v[nt];
      }
    }
    // reduce over the 16 n-lanes of each half (masks < 16 stay within half)
    #pragma unroll
    for (int r = 0; r < 8; ++r) {
      lacc[r] += __shfl_xor(lacc[r], 1, 32);
      lacc[r] += __shfl_xor(lacc[r], 2, 32);
      lacc[r] += __shfl_xor(lacc[r], 4, 32);
      lacc[r] += __shfl_xor(lacc[r], 8, 32);
    }
    if (nlo == 0) {                     // lane 0 -> rows 0-7, lane 16 -> rows 8-15
      const int pbase = t * 16 + khalf * 8;
      #pragma unroll
      for (int r = 0; r < 8; ++r) {
        const int pw = pbase + r;
        if (pw < NP) sLogit[pw] = lacc[r];
      }
    }
  }
  __syncthreads();

  // ---- Phase 3: softmax over pairs ----
  float lm = -3.402823e38f;
  for (int p = tid; p < NP; p += 256) lm = fmaxf(lm, sLogit[p]);
  sRed[tid] = lm; __syncthreads();
  for (int s = 128; s > 0; s >>= 1) {
    if (tid < s) sRed[tid] = fmaxf(sRed[tid], sRed[tid + s]);
    __syncthreads();
  }
  if (tid == 0) sM = sRed[0];
  __syncthreads();
  const float mx = sM;
  float ls = 0.0f;
  for (int p = tid; p < NP; p += 256) {
    float e = __expf(sLogit[p] - mx);
    sLogit[p] = e; ls += e;
  }
  __syncthreads();
  sRed[tid] = ls; __syncthreads();
  for (int s = 128; s > 0; s >>= 1) {
    if (tid < s) sRed[tid] += sRed[tid + s];
    __syncthreads();
  }
  if (tid == 0) sZ = sRed[0];
  __syncthreads();
  const float zval = sZ;

  // ---- Phase 4: pooled[d] = sum_p e[p]*x[i0,d]*x[i1,d]; out = (pooled/Z) . pv ----
  {
    const int dt = tid & 15;            // 16 chunks of 4 d's
    const int g  = tid >> 4;            // 16 p-groups
    const int d4 = dt * 4;
    float4 acc; acc.x = acc.y = acc.z = acc.w = 0.0f;
    for (int p = g; p < NP; p += 16) {
      const float e = sLogit[p];
      const float4 a = *reinterpret_cast<const float4*>(&sInF[sI0[p] * STRIDE + d4]);
      const float4 c = *reinterpret_cast<const float4*>(&sInF[sI1[p] * STRIDE + d4]);
      acc.x += e * (a.x * c.x);
      acc.y += e * (a.y * c.y);
      acc.z += e * (a.z * c.z);
      acc.w += e * (a.w * c.w);
    }
    *reinterpret_cast<float4*>(&sPool[g * 64 + d4]) = acc;
  }
  __syncthreads();
  if (tid < 64) {
    float pool = 0.0f;
    #pragma unroll
    for (int g = 0; g < 16; ++g) pool += sPool[g * 64 + tid];
    sRed[tid] = (pool / zval) * pv[tid];
  }
  __syncthreads();
  if (tid == 0) {
    float s = 0.0f;
    #pragma unroll
    for (int d2 = 0; d2 < 64; ++d2) s += sRed[d2];
    out[b] = s;
  }
}

extern "C" void kernel_launch(void* const* d_in, const int* in_sizes, int n_in,
                              void* d_out, int out_size, void* d_ws, size_t ws_size,
                              hipStream_t stream) {
  const float* inp = (const float*)d_in[0];   // [B,F,D]
  const float* W1  = (const float*)d_in[1];   // [D,A]
  const float* b1  = (const float*)d_in[2];   // [A]
  const float* w2  = (const float*)d_in[3];   // [A]
  const float* pv  = (const float*)d_in[4];   // [D]
  float* out = (float*)d_out;                 // [B]
  (void)in_sizes; (void)n_in; (void)out_size; (void)d_ws; (void)ws_size;
  afm_kernel<<<dim3(BATCH), dim3(256), 0, stream>>>(inp, W1, b1, w2, pv, out);
}
